// HierarchicalLocalAttention_80393197846771
// MI455X (gfx1250) — compile-verified, matrix operand order checked by reading, not run on hardware
//
#include <hip/hip_runtime.h>
#include <hip/hip_bf16.h>

typedef __bf16 bf16_t;
typedef __bf16 v16bf __attribute__((ext_vector_type(16)));
typedef __bf16 v8bf  __attribute__((ext_vector_type(8)));
typedef float  v8f   __attribute__((ext_vector_type(8)));

#define B_SZ   1024
#define C_CH   2048
#define CR_CH  128
#define HW     9
#define N_COLS (B_SZ * HW)   // 9216 columns (b,p)

// ---------------------------------------------------------------------------
// f32 -> bf16 weight conversion (once per launch; weights stay hot in L2)
// ---------------------------------------------------------------------------
__global__ void cvt_bf16_kernel(const float* __restrict__ src,
                                bf16_t* __restrict__ dst, int n) {
  int i = blockIdx.x * blockDim.x + threadIdx.x;
  if (i < n) dst[i] = (bf16_t)src[i];
}

// A-fragment for v_wmma_f32_16x16x32_bf16 (row-major bf16 weights):
// lane<16 : K = k0+0..7 (elems 0..7), k0+16..23 (elems 8..15)
// lane>=16: K = k0+8..15,             k0+24..31   (koff = 8)
__device__ inline v16bf load_a_frag(const bf16_t* __restrict__ row, int k0, int koff) {
  v8bf lo = *(const v8bf*)(row + k0 + koff);
  v8bf hh = *(const v8bf*)(row + k0 + koff + 16);
  v16bf f;
#pragma unroll
  for (int i = 0; i < 8; ++i) { f[i] = lo[i]; f[i + 8] = hh[i]; }
  return f;
}

// Stage one 32K x 32N B tile (two fragment-ordered 16-col subtiles) into LDS.
// 1024 threads -> exactly one element each, full EXEC, no divergence.
__device__ inline void stage_btile(const float* __restrict__ X, int n0, int k0,
                                   bf16_t* __restrict__ buf, int tid) {
  int kk = tid >> 5, nr = tid & 31;
  int n = n0 + nr;
  int b = n / HW, p = n % HW;
  float v = X[(size_t)b * C_CH * HW + (size_t)(k0 + kk) * HW + p];
  int nt = nr >> 4, nn = nr & 15;
  int dl = nn + (((kk >> 3) & 1) << 4);
  int de = (kk & 7) + (((kk >> 4) & 1) << 3);
  buf[nt * 512 + dl * 16 + de] = (bf16_t)v;
}

// ---------------------------------------------------------------------------
// Kernel 1: dk = softmax_channels(wd @ x + bd)   [M=K=2048, N=9216]
// One workgroup (32 waves) owns the FULL 2048-row strip for 32 columns
// (2 N-tiles), so the channel softmax reduces entirely in LDS and every A
// fragment feeds two WMMAs (halves L2 traffic for wd).
// Per wave: 4 M-tiles x 2 N-tiles. Double-buffered B tile: 1 barrier/k-step.
// ---------------------------------------------------------------------------
__global__ __launch_bounds__(1024)
void gemm_softmax_kernel(const bf16_t* __restrict__ Wbf,   // [2048,2048] bf16
                         const float* __restrict__ X,      // [B,2048,9] f32
                         const float* __restrict__ bias,   // [2048]
                         float* __restrict__ DK) {         // [B,2048,9]
  constexpr int Kdim = C_CH;
  constexpr int KSTEPS = Kdim / 32;
  const int tid  = threadIdx.x;
  const int wave = tid >> 5;
  const int lane = tid & 31;
  const int hi   = lane >> 4;
  const int col  = lane & 15;
  const int koff = hi ? 8 : 0;
  const int n0   = blockIdx.x * 32;

  __shared__ __align__(64) bf16_t bTile[2][2 * 512]; // double-buffered
  __shared__ float red[64 * 32];
  __shared__ float cstat[32];

  v8f acc[2][4];
#pragma unroll
  for (int nt = 0; nt < 2; ++nt)
#pragma unroll
    for (int t = 0; t < 4; ++t)
      acc[nt][t] = (v8f){0.f,0.f,0.f,0.f,0.f,0.f,0.f,0.f};

  stage_btile(X, n0, 0, bTile[0], tid);
  __syncthreads();

  for (int ks = 0; ks < KSTEPS; ++ks) {
    const int k0 = ks * 32;
    const bf16_t* cur = bTile[ks & 1];
    v16bf bfrag0 = *(const v16bf*)(cur + lane * 16);
    v16bf bfrag1 = *(const v16bf*)(cur + 512 + lane * 16);
    if (ks + 1 < KSTEPS)
      stage_btile(X, n0, k0 + 32, bTile[(ks + 1) & 1], tid);
    __syncthreads();
#pragma unroll
    for (int t = 0; t < 4; ++t) {
      int m = wave * 64 + t * 16 + col;
      v16bf afrag = load_a_frag(Wbf + (size_t)m * Kdim, k0, koff);
      acc[0][t] = __builtin_amdgcn_wmma_f32_16x16x32_bf16(
          false, afrag, false, bfrag0, (short)0, acc[0][t], false, false);
      acc[1][t] = __builtin_amdgcn_wmma_f32_16x16x32_bf16(
          false, afrag, false, bfrag1, (short)0, acc[1][t], false, false);
    }
  }

  // bias + channel softmax across all 2048 rows of the 32-column strip
  float pmax[2] = {-3.402823466e38f, -3.402823466e38f};
#pragma unroll
  for (int t = 0; t < 4; ++t)
#pragma unroll
    for (int v = 0; v < 8; ++v) {
      int m = wave * 64 + t * 16 + v + 8 * hi;
      float bv = bias[m];
#pragma unroll
      for (int nt = 0; nt < 2; ++nt) {
        acc[nt][t][v] += bv;
        pmax[nt] = fmaxf(pmax[nt], acc[nt][t][v]);
      }
    }
  const int grp = wave * 2 + hi;           // 64 groups x 32 columns
  red[grp * 32 + col]      = pmax[0];
  red[grp * 32 + col + 16] = pmax[1];
  __syncthreads();
  if (tid < 32) {
    float m = red[tid];
    for (int i = 1; i < 64; ++i) m = fmaxf(m, red[i * 32 + tid]);
    cstat[tid] = m;
  }
  __syncthreads();
  float cmax[2] = {cstat[col], cstat[col + 16]};
  float psum[2] = {0.f, 0.f};
#pragma unroll
  for (int nt = 0; nt < 2; ++nt)
#pragma unroll
    for (int t = 0; t < 4; ++t)
#pragma unroll
      for (int v = 0; v < 8; ++v) {
        acc[nt][t][v] = __expf(acc[nt][t][v] - cmax[nt]);
        psum[nt] += acc[nt][t][v];
      }
  red[grp * 32 + col]      = psum[0];
  red[grp * 32 + col + 16] = psum[1];
  __syncthreads();
  if (tid < 32) {
    float s = 0.f;
    for (int i = 0; i < 64; ++i) s += red[i * 32 + tid];
    cstat[tid] = 1.0f / s;
  }
  __syncthreads();
  float rinv[2] = {cstat[col], cstat[col + 16]};
#pragma unroll
  for (int nt = 0; nt < 2; ++nt) {
    const int n = n0 + nt * 16 + col;
    const int nb = n / HW, np = n % HW;
    const size_t outbase = (size_t)nb * C_CH * HW + np;
#pragma unroll
    for (int t = 0; t < 4; ++t)
#pragma unroll
      for (int v = 0; v < 8; ++v) {
        int m = wave * 64 + t * 16 + v + 8 * hi;
        DK[outbase + (size_t)m * HW] = acc[nt][t][v] * rinv[nt];
      }
  }
}

// ---------------------------------------------------------------------------
// Generic column GEMM:  Y[b,m,p] = act(W[m,:] . X[b,:,p] + bias[m])
// M = WAVES*TILES*16, grid.x = N_COLS/16
// ---------------------------------------------------------------------------
template <int WAVES, int TILES, bool SIG>
__global__ void gemm_cols_kernel(const bf16_t* __restrict__ Wbf,
                                 const float* __restrict__ X,
                                 const float* __restrict__ bias,
                                 float* __restrict__ Y,
                                 int Mdim, int Kdim) {
  const int tid  = threadIdx.x;
  const int wave = tid >> 5;
  const int lane = tid & 31;
  const int hi   = lane >> 4;
  const int col  = lane & 15;
  const int koff = hi ? 8 : 0;
  const int n0   = blockIdx.x * 16;

  __shared__ __align__(64) bf16_t bTile[32 * 16];

  v8f acc[TILES];
#pragma unroll
  for (int t = 0; t < TILES; ++t) acc[t] = (v8f){0.f,0.f,0.f,0.f,0.f,0.f,0.f,0.f};

  for (int ks = 0; ks < Kdim / 32; ++ks) {
    const int k0 = ks * 32;
    for (int idx = tid; idx < 512; idx += WAVES * 32) {
      int kk = idx >> 4, nr = idx & 15;
      int n = n0 + nr;
      int b = n / HW, p = n % HW;
      float v = X[(size_t)b * Kdim * HW + (size_t)(k0 + kk) * HW + p];
      int dl = nr + (((kk >> 3) & 1) << 4);
      int de = (kk & 7) + (((kk >> 4) & 1) << 3);
      bTile[dl * 16 + de] = (bf16_t)v;
    }
    __syncthreads();
    v16bf bfrag = *(const v16bf*)(bTile + lane * 16);
#pragma unroll
    for (int t = 0; t < TILES; ++t) {
      int m = (wave * TILES + t) * 16 + col;
      v16bf afrag = load_a_frag(Wbf + (size_t)m * Kdim, k0, koff);
      acc[t] = __builtin_amdgcn_wmma_f32_16x16x32_bf16(
          false, afrag, false, bfrag, (short)0, acc[t], false, false);
    }
    __syncthreads();
  }

  const int n = n0 + col;
  const int nb = n / HW, np = n % HW;
#pragma unroll
  for (int t = 0; t < TILES; ++t)
#pragma unroll
    for (int v = 0; v < 8; ++v) {
      int m = (wave * TILES + t) * 16 + v + 8 * hi;
      float y = acc[t][v] + bias[m];
      if (SIG) y = 1.0f / (1.0f + __expf(-y));
      Y[(size_t)nb * Mdim * HW + (size_t)m * HW + np] = y;
    }
}

// ---------------------------------------------------------------------------
// Kernel 3: 3x depthwise 3x3 'same' conv chain on [B,128,3,3]
// ---------------------------------------------------------------------------
__device__ inline void dw3x3(const float* in, const float* __restrict__ w,
                             float bias, float* out) {
#pragma unroll
  for (int i = 0; i < 3; ++i)
#pragma unroll
    for (int j = 0; j < 3; ++j) {
      float a = bias;
#pragma unroll
      for (int di = 0; di < 3; ++di)
#pragma unroll
        for (int dj = 0; dj < 3; ++dj) {
          int ii = i + di - 1, jj = j + dj - 1;
          if (ii >= 0 && ii < 3 && jj >= 0 && jj < 3)
            a += in[ii * 3 + jj] * w[di * 3 + dj];
        }
      out[i * 3 + j] = a;
    }
}

__global__ void dwchain_kernel(const float* __restrict__ S,
                               const float* __restrict__ wl1, const float* __restrict__ bl1,
                               const float* __restrict__ wl2, const float* __restrict__ bl2,
                               const float* __restrict__ wl3, const float* __restrict__ bl3,
                               float* __restrict__ T) {
  int idx = blockIdx.x * blockDim.x + threadIdx.x;
  if (idx >= B_SZ * CR_CH) return;
  int c = idx % CR_CH;
  float a[9], t1[9], t2[9];
#pragma unroll
  for (int i = 0; i < 9; ++i) a[i] = S[(size_t)idx * 9 + i];
  dw3x3(a,  wl1 + c * 9, bl1[c], t1);
  dw3x3(t1, wl2 + c * 9, bl2[c], t2);
  dw3x3(t2, wl3 + c * 9, bl3[c], a);
#pragma unroll
  for (int i = 0; i < 9; ++i) T[(size_t)idx * 9 + i] = a[i];
}

// ---------------------------------------------------------------------------
// Kernel 5: att[b,c,i,j] = sum_{a,b'} pad(g)[i+a,j+b'] * dk[a,b'];
// out = x * att.  dk lives in d_out and is overwritten in place (thread-local).
// ---------------------------------------------------------------------------
__global__ void attn_apply_kernel(const float* __restrict__ X,
                                  const float* __restrict__ G,
                                  float* __restrict__ OUT) {
  int idx = blockIdx.x * blockDim.x + threadIdx.x;
  if (idx >= B_SZ * C_CH) return;
  const size_t base = (size_t)idx * 9;
  float g[9], dk[9], xv[9], o[9];
#pragma unroll
  for (int i = 0; i < 9; ++i) { g[i] = G[base + i]; dk[i] = OUT[base + i]; xv[i] = X[base + i]; }
#pragma unroll
  for (int i = 0; i < 3; ++i)
#pragma unroll
    for (int j = 0; j < 3; ++j) {
      float a = 0.f;
#pragma unroll
      for (int di = 0; di < 3; ++di)
#pragma unroll
        for (int dj = 0; dj < 3; ++dj) {
          int ii = i + di - 1, jj = j + dj - 1;
          if (ii >= 0 && ii < 3 && jj >= 0 && jj < 3)
            a += g[ii * 3 + jj] * dk[di * 3 + dj];
        }
      o[i * 3 + j] = xv[i * 3 + j] * a;
    }
#pragma unroll
  for (int i = 0; i < 9; ++i) OUT[base + i] = o[i];
}

// ---------------------------------------------------------------------------
extern "C" void kernel_launch(void* const* d_in, const int* in_sizes, int n_in,
                              void* d_out, int out_size, void* d_ws, size_t ws_size,
                              hipStream_t stream) {
  const float* x   = (const float*)d_in[0];
  const float* w1  = (const float*)d_in[1];
  const float* b1  = (const float*)d_in[2];
  const float* wl1 = (const float*)d_in[3];
  const float* bl1 = (const float*)d_in[4];
  const float* wl2 = (const float*)d_in[5];
  const float* bl2 = (const float*)d_in[6];
  const float* wl3 = (const float*)d_in[7];
  const float* bl3 = (const float*)d_in[8];
  const float* w2  = (const float*)d_in[9];
  const float* b2  = (const float*)d_in[10];
  const float* wd  = (const float*)d_in[11];
  const float* bd  = (const float*)d_in[12];
  float* out = (float*)d_out;

  // workspace: bf16 weights + squeeze/dw/gate buffers (~94 MB)
  bf16_t* wdbf = (bf16_t*)d_ws;
  bf16_t* w1bf = wdbf + (size_t)C_CH * C_CH;
  bf16_t* w2bf = w1bf + (size_t)CR_CH * C_CH;
  float*  sbuf = (float*)(w2bf + (size_t)C_CH * CR_CH);
  float*  tbuf = sbuf + (size_t)B_SZ * CR_CH * HW;
  float*  gbuf = tbuf + (size_t)B_SZ * CR_CH * HW;

  // 0) weight conversions
  cvt_bf16_kernel<<<(C_CH * C_CH + 255) / 256, 256, 0, stream>>>(wd, wdbf, C_CH * C_CH);
  cvt_bf16_kernel<<<(CR_CH * C_CH + 255) / 256, 256, 0, stream>>>(w1, w1bf, CR_CH * C_CH);
  cvt_bf16_kernel<<<(C_CH * CR_CH + 255) / 256, 256, 0, stream>>>(w2, w2bf, C_CH * CR_CH);

  // 1) dk = softmax(wd @ x + bd) -> d_out  (77 GFLOP WMMA GEMM + fused softmax)
  gemm_softmax_kernel<<<N_COLS / 32, 1024, 0, stream>>>(wdbf, x, bd, out);

  // 2) squeeze: s = w1 @ x + b1  (M=128)
  gemm_cols_kernel<8, 1, false><<<N_COLS / 16, 256, 0, stream>>>(w1bf, x, b1, sbuf, CR_CH, C_CH);

  // 3) depthwise 3x3 chain
  dwchain_kernel<<<(B_SZ * CR_CH + 255) / 256, 256, 0, stream>>>(
      sbuf, wl1, bl1, wl2, bl2, wl3, bl3, tbuf);

  // 4) expand + sigmoid: g = sigmoid(w2 @ t + b2)  (M=2048, K=128)
  gemm_cols_kernel<16, 8, true><<<N_COLS / 16, 512, 0, stream>>>(w2bf, tbuf, b2, gbuf, C_CH, CR_CH);

  // 5) att = corr3x3(pad(g), dk); out = x * att  (dk in d_out, overwritten)
  attn_apply_kernel<<<(B_SZ * C_CH + 255) / 256, 256, 0, stream>>>(x, gbuf, out);
}
